// PolymerGNN_69741678953162
// MI455X (gfx1250) — compile-verified, matrix operand order checked by reading, not run on hardware
//
#include <hip/hip_runtime.h>

#define DEV __device__ __forceinline__

// ---------------- problem constants (from reference) ----------------
constexpr int Nn  = 51200;           // nodes
constexpr int Gg  = 512;             // graphs
constexpr int NPGc = 100;            // nodes per graph
constexpr int Ee  = 300000;          // edges
constexpr int Hh  = 256;             // hidden
constexpr int NHc = 8;               // heads
constexpr int DHc = 32;              // head dim
constexpr int Lc  = 4;               // GAT layers
constexpr int Tc  = 5;               // tasks
constexpr int EF  = Ee + Nn;         // edges + self loops
constexpr size_t NHF = (size_t)Nn * Hh;

// ---------------- WMMA types ----------------
typedef __attribute__((ext_vector_type(16))) __bf16 v16bf;
typedef __attribute__((ext_vector_type(8)))  __bf16 v8bf;
typedef __attribute__((ext_vector_type(8)))  float  v8f;

// ordered-uint encoding for float atomicMax
DEV unsigned f_ord(float f) {
  unsigned u = __float_as_uint(f);
  return (u & 0x80000000u) ? ~u : (u | 0x80000000u);
}
DEV float f_unord(unsigned e) {
  unsigned u = (e & 0x80000000u) ? (e ^ 0x80000000u) : ~e;
  return __uint_as_float(u);
}

DEV v16bf cat16(v8bf lo, v8bf hi) {
  return __builtin_shufflevector(lo, hi, 0, 1, 2, 3, 4, 5, 6, 7,
                                 8, 9, 10, 11, 12, 13, 14, 15);
}

DEV v16bf load_frag(const __bf16* p) {
  return cat16(*(const v8bf*)p, *(const v8bf*)(p + 16));
}

// ---------------- bf16 WMMA GEMM (pre-converted operands) ----------------
// C[M,Nc] = act( A @ B + bias ), A: bf16 [M,K] row-major, BT: bf16 [Nc,K]
// row-major (i.e. B transposed).  Each wave computes a 32x32 output slab:
// 2 A fragments x 2 B fragments -> 4 accumulators (1 wmma per fragment load).
// K templated so the loop fully unrolls into back-to-back v_wmma.
template <int K>
__global__ __launch_bounds__(256)
void k_wmma_gemm(const __bf16* __restrict__ A, const __bf16* __restrict__ BT,
                 const float* __restrict__ bias, float* __restrict__ C,
                 __bf16* __restrict__ Cbf, int M, int Nc, int act)
{
  const int lane = threadIdx.x & 31;
  const int wave = threadIdx.x >> 5;
  const int colPairs = Nc >> 5;
  const int rowPairs = M >> 5;
  const int totalTiles = rowPairs * colPairs;
  const int tile = blockIdx.x * 8 + wave;
  if (tile >= totalTiles) return;
  const int mt = tile / colPairs;
  const int np = tile - mt * colPairs;
  const int r = lane & 15;          // A row / B,C col within 16-tile
  const int half = lane >> 4;       // K-half selector per ISA layout
  const int ar0 = (mt << 5) + r;    // A rows
  const int ar1 = ar0 + 16;
  const int c0 = (np << 5) + r;     // output columns
  const int c1 = c0 + 16;

  v8f acc00 = {0.f, 0.f, 0.f, 0.f, 0.f, 0.f, 0.f, 0.f};
  v8f acc01 = acc00, acc10 = acc00, acc11 = acc00;

  const __bf16* Ap0 = A  + (size_t)ar0 * K + half * 8;
  const __bf16* Ap1 = A  + (size_t)ar1 * K + half * 8;
  const __bf16* Bp0 = BT + (size_t)c0 * K + half * 8;
  const __bf16* Bp1 = BT + (size_t)c1 * K + half * 8;

#pragma unroll
  for (int k0 = 0; k0 < K; k0 += 32) {
    __builtin_prefetch(Ap0 + k0 + 64, 0, 1);
    v16bf a0 = load_frag(Ap0 + k0);
    v16bf a1 = load_frag(Ap1 + k0);
    v16bf b0 = load_frag(Bp0 + k0);
    v16bf b1 = load_frag(Bp1 + k0);
    acc00 = __builtin_amdgcn_wmma_f32_16x16x32_bf16(false, a0, false, b0,
                                                    (short)0, acc00, false, false);
    acc01 = __builtin_amdgcn_wmma_f32_16x16x32_bf16(false, a0, false, b1,
                                                    (short)0, acc01, false, false);
    acc10 = __builtin_amdgcn_wmma_f32_16x16x32_bf16(false, a1, false, b0,
                                                    (short)0, acc10, false, false);
    acc11 = __builtin_amdgcn_wmma_f32_16x16x32_bf16(false, a1, false, b1,
                                                    (short)0, acc11, false, false);
  }

  const float bv0 = bias ? bias[c0] : 0.f;
  const float bv1 = bias ? bias[c1] : 0.f;
#pragma unroll
  for (int v = 0; v < 8; ++v) {
    const int m0 = (mt << 5) + (half << 3) + v;   // D layout: M = v + 8*half
    const int m1 = m0 + 16;
    float x00 = acc00[v] + bv0, x01 = acc01[v] + bv1;
    float x10 = acc10[v] + bv0, x11 = acc11[v] + bv1;
    if (act == 1) {
      x00 = fmaxf(x00, 0.f); x01 = fmaxf(x01, 0.f);
      x10 = fmaxf(x10, 0.f); x11 = fmaxf(x11, 0.f);
    }
    C[(size_t)m0 * Nc + c0] = x00;
    C[(size_t)m0 * Nc + c1] = x01;
    C[(size_t)m1 * Nc + c0] = x10;
    C[(size_t)m1 * Nc + c1] = x11;
    if (Cbf) {
      Cbf[(size_t)m0 * Nc + c0] = (__bf16)x00;
      Cbf[(size_t)m0 * Nc + c1] = (__bf16)x01;
      Cbf[(size_t)m1 * Nc + c0] = (__bf16)x10;
      Cbf[(size_t)m1 * Nc + c1] = (__bf16)x11;
    }
  }
}

// ---------------- conversion kernels ----------------
__global__ void k_cvt(const float* __restrict__ src, __bf16* __restrict__ dst, int n) {
  int t = blockIdx.x * blockDim.x + threadIdx.x;
  if (t < n) dst[t] = (__bf16)src[t];
}
// [K,Nc] f32 -> [Nc,K] bf16 (weight transpose for the GEMM B layout)
__global__ void k_cvt_T(const float* __restrict__ src, __bf16* __restrict__ dst,
                        int K, int Nc) {
  int t = blockIdx.x * blockDim.x + threadIdx.x;
  if (t >= K * Nc) return;
  int k = t / Nc, n = t - k * Nc;
  dst[(size_t)n * K + k] = (__bf16)src[t];
}

// ---------------- small utility kernels ----------------
__global__ void k_zero(float* p, int n) {
  int t = blockIdx.x * blockDim.x + threadIdx.x;
  if (t < n) p[t] = 0.f;
}

// segment-sum of edge_attr by dst + degree
__global__ void k_edge_accum(const int* __restrict__ ei, const float* __restrict__ ea,
                             float* __restrict__ ea_sum, float* __restrict__ deg)
{
  int j = blockIdx.x * blockDim.x + threadIdx.x;
  if (j >= Ee) return;
  int d = ei[Ee + j];
  atomicAdd(&deg[d], 1.f);
#pragma unroll
  for (int u = 0; u < 8; ++u)
    atomicAdd(&ea_sum[d * 8 + u], ea[(size_t)j * 8 + u]);
}

__global__ void k_ea_mean(const float* __restrict__ ea_sum, const float* __restrict__ deg,
                          float* __restrict__ ea_mean)
{
  int t = blockIdx.x * blockDim.x + threadIdx.x;
  if (t >= Nn * 8) return;
  int n = t >> 3;
  ea_mean[t] = ea_sum[t] / fmaxf(deg[n], 1.f);
}

// Fold edge path: K8[f][h] = edge_w @ (W_e @ att_e) ; c[h] = edge_b @ (W_e @ att_e)
__global__ __launch_bounds__(256)
void k_fold_edge(const float* __restrict__ lw_e,   // [H,H] layer slice
                 const float* __restrict__ att_e,  // [NH*DH] layer slice
                 const float* __restrict__ edge_w, // [8,H]
                 const float* __restrict__ edge_b, // [H]
                 float* __restrict__ K8c)          // [72]: K8 [8][8] then c[8]
{
  __shared__ float Ms[Hh * NHc];
  int t = threadIdx.x;  // 256 = one k per thread
#pragma unroll
  for (int hd = 0; hd < NHc; ++hd) {
    float acc = 0.f;
#pragma unroll
    for (int dd = 0; dd < DHc; ++dd)
      acc += lw_e[(size_t)t * Hh + hd * DHc + dd] * att_e[hd * DHc + dd];
    Ms[t * NHc + hd] = acc;
  }
  __syncthreads();
  if (t < 64) {
    int f = t >> 3, hd = t & 7;
    float acc = 0.f;
    for (int k = 0; k < Hh; ++k) acc += edge_w[f * Hh + k] * Ms[k * NHc + hd];
    K8c[f * 8 + hd] = acc;
  } else if (t < 72) {
    int hd = t - 64;
    float acc = 0.f;
    for (int k = 0; k < Hh; ++k) acc += edge_b[k] * Ms[k * NHc + hd];
    K8c[64 + hd] = acc;
  }
}

// a_src/a_dst per (node, head) from xs
__global__ void k_asrc_adst(const float* __restrict__ xs,
                            const float* __restrict__ att_s, const float* __restrict__ att_d,
                            float* __restrict__ a_src, float* __restrict__ a_dst)
{
  int t = blockIdx.x * blockDim.x + threadIdx.x;
  if (t >= Nn * NHc) return;
  int n = t >> 3, hd = t & 7;
  const float* xp = xs + (size_t)n * Hh + hd * DHc;
  float as = 0.f, ad = 0.f;
#pragma unroll
  for (int d = 0; d < DHc; ++d) {
    as += xp[d] * att_s[hd * DHc + d];
    ad += xp[d] * att_d[hd * DHc + d];
  }
  a_src[t] = as;
  a_dst[t] = ad;
}

// attention logits (leaky relu) + segment max over dst
__global__ void k_alpha(const int* __restrict__ ei, const float* __restrict__ ea,
                        const float* __restrict__ ea_mean, const float* __restrict__ deg,
                        const float* __restrict__ K8c,
                        const float* __restrict__ a_src, const float* __restrict__ a_dst,
                        float* __restrict__ alpha, unsigned* __restrict__ mx)
{
  int j = blockIdx.x * blockDim.x + threadIdx.x;
  if (j >= EF) return;
  float f[8];
  int s, d;
  float cscale;
  if (j < Ee) {
    s = ei[j]; d = ei[Ee + j];
    cscale = 1.f;
#pragma unroll
    for (int u = 0; u < 8; ++u) f[u] = ea[(size_t)j * 8 + u];
  } else {
    int n = j - Ee;
    s = d = n;
    cscale = (deg[n] > 0.f) ? 1.f : 0.f;    // e_self bias term vanishes for deg==0
#pragma unroll
    for (int u = 0; u < 8; ++u) f[u] = ea_mean[n * 8 + u];
  }
#pragma unroll
  for (int hd = 0; hd < NHc; ++hd) {
    float aev = cscale * K8c[64 + hd];
#pragma unroll
    for (int u = 0; u < 8; ++u) aev += f[u] * K8c[u * 8 + hd];
    float a = a_src[s * NHc + hd] + a_dst[d * NHc + hd] + aev;
    a = (a > 0.f) ? a : 0.2f * a;           // leaky relu 0.2
    alpha[(size_t)j * NHc + hd] = a;
    atomicMax(&mx[d * NHc + hd], f_ord(a));
  }
}

// ex = exp(alpha - mx[dst]); den[dst] += ex  (in-place alpha -> ex)
__global__ void k_expsum(const int* __restrict__ ei, float* __restrict__ exb,
                         const unsigned* __restrict__ mx, float* __restrict__ den)
{
  int t = blockIdx.x * blockDim.x + threadIdx.x;
  if (t >= EF * NHc) return;
  int j = t >> 3, hd = t & 7;
  int d = (j < Ee) ? ei[Ee + j] : (j - Ee);
  float m = f_unord(mx[d * NHc + hd]);
  float e = __expf(exb[t] - m);
  exb[t] = e;
  atomicAdd(&den[d * NHc + hd], e);
}

// out[dst] += xs[src] * w   (4 channels per thread)
__global__ void k_aggregate(const int* __restrict__ ei, const float* __restrict__ xs,
                            const float* __restrict__ exb, const float* __restrict__ den,
                            float* __restrict__ out)
{
  int t = blockIdx.x * blockDim.x + threadIdx.x;
  if (t >= EF * 64) return;
  int j  = t >> 6;
  int c0 = (t & 63) * 4;
  int hd = c0 >> 5;
  int s, d;
  if (j < Ee) { s = ei[j]; d = ei[Ee + j]; } else { s = d = j - Ee; }
  float w = exb[(size_t)j * NHc + hd] / (den[d * NHc + hd] + 1e-16f);
  const float* xp = xs + (size_t)s * Hh + c0;
  float* op = out + (size_t)d * Hh + c0;
#pragma unroll
  for (int u = 0; u < 4; ++u) atomicAdd(op + u, xp[u] * w);
}

// per-channel sum / sumsq (training-mode BN stats)
__global__ __launch_bounds__(256)
void k_bn_stats(const float* __restrict__ out, float* __restrict__ bsum, float* __restrict__ bss)
{
  int c = threadIdx.x;               // channel
  int r0 = blockIdx.x * 128;         // row chunk
  float s = 0.f, s2 = 0.f;
  for (int r = 0; r < 128; ++r) {
    float v = out[(size_t)(r0 + r) * Hh + c];
    s += v; s2 += v * v;
  }
  atomicAdd(&bsum[c], s);
  atomicAdd(&bss[c], s2);
}

// h = relu(BN(out)) (+ residual); also emits bf16 copy for the next GEMM
__global__ void k_bn_apply(const float* __restrict__ out,
                           const float* __restrict__ bsum, const float* __restrict__ bss,
                           const float* __restrict__ gamma, const float* __restrict__ beta,
                           float* __restrict__ h, __bf16* __restrict__ hbf, int addres)
{
  int t = blockIdx.x * blockDim.x + threadIdx.x;
  if (t >= (int)NHF) return;
  int c = t & (Hh - 1);
  const float invN = 1.f / (float)Nn;
  float mu  = bsum[c] * invN;
  float var = bss[c] * invN - mu * mu;
  float v = gamma[c] * (out[t] - mu) * rsqrtf(var + 1e-5f) + beta[c];
  v = fmaxf(v, 0.f);
  v = v + (addres ? h[t] : 0.f);
  h[t] = v;
  hbf[t] = (__bf16)v;
}

// per-(graph,head) attention; K staged in LDS, V read through cache
__global__ __launch_bounds__(128)
void k_mha_attn(const float* __restrict__ qkv, float* __restrict__ o,
                __bf16* __restrict__ obf)
{
  __shared__ float ks[NPGc * DHc];   // 3200 floats = 12.5 KB
  int g  = blockIdx.x >> 3;
  int hd = blockIdx.x & 7;
  const float* base = qkv + (size_t)g * NPGc * (3 * Hh);
  for (int idx = threadIdx.x; idx < NPGc * DHc; idx += 128) {
    int rr = idx >> 5, dd = idx & 31;
    ks[idx] = base[(size_t)rr * (3 * Hh) + Hh + hd * DHc + dd];
  }
  __syncthreads();
  int i = threadIdx.x;
  if (i >= NPGc) return;
  const float scale = 0.17677669529663687f;  // 1/sqrt(32)
  float q[DHc];
  const float* qp = base + (size_t)i * (3 * Hh) + hd * DHc;
#pragma unroll
  for (int d = 0; d < DHc; ++d) q[d] = qp[d] * scale;

  float m = -1e30f;
  for (int j = 0; j < NPGc; ++j) {
    float sdot = 0.f;
#pragma unroll
    for (int d = 0; d < DHc; ++d) sdot += q[d] * ks[j * DHc + d];
    m = fmaxf(m, sdot);
  }
  float denv = 0.f;
  float ov[DHc];
#pragma unroll
  for (int d = 0; d < DHc; ++d) ov[d] = 0.f;
  for (int j = 0; j < NPGc; ++j) {
    float sdot = 0.f;
#pragma unroll
    for (int d = 0; d < DHc; ++d) sdot += q[d] * ks[j * DHc + d];
    float e = __expf(sdot - m);
    denv += e;
    const float* vp = base + (size_t)j * (3 * Hh) + 2 * Hh + hd * DHc;
#pragma unroll
    for (int d = 0; d < DHc; ++d) ov[d] += e * vp[d];
  }
  float inv = 1.f / denv;
  size_t ob = ((size_t)g * NPGc + i) * Hh + hd * DHc;
#pragma unroll
  for (int d = 0; d < DHc; ++d) {
    float val = ov[d] * inv;
    o[ob + d] = val;
    obf[ob + d] = (__bf16)val;
  }
}

// mean pool over nodes of each graph (+ bf16 copy)
__global__ __launch_bounds__(256)
void k_pool(const float* __restrict__ proj, float* __restrict__ pooled,
            __bf16* __restrict__ pbf)
{
  int g = blockIdx.x, c = threadIdx.x;
  float s = 0.f;
  for (int i = 0; i < NPGc; ++i) s += proj[((size_t)g * NPGc + i) * Hh + c];
  float v = s * (1.f / (float)NPGc);
  pooled[(size_t)g * Hh + c] = v;
  pbf[(size_t)g * Hh + c] = (__bf16)v;
}

// pred[g,t] = hh[t][g] . head_w2[t] + head_b2[t]
__global__ void k_head_final(const float* __restrict__ hh, const float* __restrict__ w2,
                             const float* __restrict__ b2, float* __restrict__ outp)
{
  int t = blockIdx.x * blockDim.x + threadIdx.x;
  if (t >= Gg * Tc) return;
  int g = t / Tc, tk = t - g * Tc;
  const float* hp = hh + ((size_t)tk * Gg + g) * 64;
  const float* wp = w2 + tk * 64;
  float s = b2[tk];
#pragma unroll
  for (int k = 0; k < 64; ++k) s += hp[k] * wp[k];
  outp[(size_t)g * Tc + tk] = s;
}

// ---------------- host-side launch helpers ----------------
static inline void gemm(const __bf16* A, const __bf16* BT, const float* bias,
                        float* C, __bf16* Cbf, int M, int K, int Nc, int act,
                        hipStream_t s)
{
  int tiles = (M / 32) * (Nc / 32);
  int blocks = (tiles + 7) / 8;
  switch (K) {
    case 64:  k_wmma_gemm<64> <<<blocks, 256, 0, s>>>(A, BT, bias, C, Cbf, M, Nc, act); break;
    case 128: k_wmma_gemm<128><<<blocks, 256, 0, s>>>(A, BT, bias, C, Cbf, M, Nc, act); break;
    default:  k_wmma_gemm<256><<<blocks, 256, 0, s>>>(A, BT, bias, C, Cbf, M, Nc, act); break;
  }
}
static inline void zero(float* p, long n, hipStream_t s)
{
  k_zero<<<(int)((n + 255) / 256), 256, 0, s>>>(p, (int)n);
}
static inline void cvt(const float* src, __bf16* dst, long n, hipStream_t s)
{
  k_cvt<<<(int)((n + 255) / 256), 256, 0, s>>>(src, dst, (int)n);
}
static inline void cvtT(const float* src, __bf16* dst, int K, int Nc, hipStream_t s)
{
  k_cvt_T<<<(K * Nc + 255) / 256, 256, 0, s>>>(src, dst, K, Nc);
}

extern "C" void kernel_launch(void* const* d_in, const int* in_sizes, int n_in,
                              void* d_out, int out_size, void* d_ws, size_t ws_size,
                              hipStream_t stream)
{
  const float* x        = (const float*)d_in[0];
  const float* ea       = (const float*)d_in[1];
  const int*   ei       = (const int*)d_in[2];
  // d_in[3]: batch (unused; fixed-size graphs)
  const float* atom_w   = (const float*)d_in[4];
  const float* atom_b   = (const float*)d_in[5];
  const float* edge_w   = (const float*)d_in[6];
  const float* edge_b   = (const float*)d_in[7];
  const float* lin_w    = (const float*)d_in[8];
  const float* lin_ew   = (const float*)d_in[9];
  const float* att_s    = (const float*)d_in[10];
  const float* att_d    = (const float*)d_in[11];
  const float* att_e    = (const float*)d_in[12];
  // d_in[13]: gat_bias (mathematically cancels through training-mode BN)
  const float* bn_g     = (const float*)d_in[14];
  const float* bn_b     = (const float*)d_in[15];
  const float* mha_in_w = (const float*)d_in[16];
  const float* mha_in_b = (const float*)d_in[17];
  const float* mha_ow   = (const float*)d_in[18];
  const float* mha_ob   = (const float*)d_in[19];
  const float* sh_w1    = (const float*)d_in[20];
  const float* sh_b1    = (const float*)d_in[21];
  const float* sh_w2    = (const float*)d_in[22];
  const float* sh_b2    = (const float*)d_in[23];
  const float* head_w1  = (const float*)d_in[24];
  const float* head_b1  = (const float*)d_in[25];
  const float* head_w2  = (const float*)d_in[26];
  const float* head_b2  = (const float*)d_in[27];

  float* ws = (float*)d_ws;
  // big f32 buffers (reuse: GAT 'out' + 'xs' become the qkv region, then proj)
  float* h    = ws;                    // [N,H]
  float* qkv  = ws + 1 * NHF;          // [N,3H] = q|k|v per row; region reused by out/proj, xs
  float* xs   = ws + 2 * NHF;          // GAT xs (inside the qkv region, overwritten later)
  float* obuf = ws + 4 * NHF;          // attention output [N,H]
  float* gout = qkv;                   // GAT aggregate output / proj output
  // bf16 buffers (element counts; 2 per float slot)
  float* sm = ws + 5 * NHF;
  __bf16* hbf = (__bf16*)sm;           sm += NHF / 2;            // [N,H] bf16
  __bf16* obf = (__bf16*)sm;           sm += NHF / 2;            // [N,H] bf16
  __bf16* xbf = (__bf16*)sm;           sm += (size_t)Nn * 64 / 2;
  __bf16* wbf = (__bf16*)sm;           sm += (size_t)768 * 256 / 2;
  __bf16* pbf = (__bf16*)sm;           sm += (size_t)Gg * Hh / 2;
  __bf16* s1bf = (__bf16*)sm;          sm += (size_t)Gg * Hh / 2;
  __bf16* s2bf = (__bf16*)sm;          sm += (size_t)Gg * (Hh / 2) / 2;
  // small f32 buffers
  float* ea_sum  = sm;                 sm += (size_t)Nn * 8;
  float* deg     = sm;                 sm += Nn;
  float* ea_mean = sm;                 sm += (size_t)Nn * 8;
  float* a_src   = sm;                 sm += (size_t)Nn * 8;
  float* a_dst   = sm;                 sm += (size_t)Nn * 8;
  float* mxf     = sm;                 sm += (size_t)Nn * 8;   // ordered-uint max
  float* den     = sm;                 sm += (size_t)Nn * 8;
  float* exb     = sm;                 sm += (size_t)EF * 8;   // alpha, then ex
  float* bnsum   = sm;                 sm += 256;
  float* bnss    = sm;                 sm += 256;
  float* K8c     = sm;                 sm += 128;
  float* pooled  = sm;                 sm += (size_t)Gg * Hh;
  float* s1      = sm;                 sm += (size_t)Gg * Hh;
  float* s2      = sm;                 sm += (size_t)Gg * (Hh / 2);
  float* hh      = sm;                 sm += (size_t)Tc * Gg * 64;

  // ---- edge preprocessing: segment mean of edge_attr by dst ----
  zero(ea_sum, (long)Nn * 9, stream);  // ea_sum + deg (contiguous)
  k_edge_accum<<<(Ee + 255) / 256, 256, 0, stream>>>(ei, ea, ea_sum, deg);
  k_ea_mean<<<(Nn * 8 + 255) / 256, 256, 0, stream>>>(ea_sum, deg, ea_mean);

  // ---- atom encoder: h = x @ atom_w + atom_b ----
  cvt(x, xbf, (long)Nn * 64, stream);
  cvtT(atom_w, wbf, 64, Hh, stream);
  gemm(xbf, wbf, atom_b, h, hbf, Nn, 64, Hh, 0, stream);

  // ---- GAT layers ----
  for (int i = 0; i < Lc; ++i) {
    k_fold_edge<<<1, 256, 0, stream>>>(lin_ew + (size_t)i * Hh * Hh,
                                       att_e + (size_t)i * Hh,
                                       edge_w, edge_b, K8c);
    cvtT(lin_w + (size_t)i * Hh * Hh, wbf, Hh, Hh, stream);
    gemm(hbf, wbf, nullptr, xs, nullptr, Nn, Hh, Hh, 0, stream);
    k_asrc_adst<<<(Nn * 8 + 255) / 256, 256, 0, stream>>>(
        xs, att_s + (size_t)i * Hh, att_d + (size_t)i * Hh, a_src, a_dst);
    zero(gout, (long)NHF, stream);
    zero(mxf, (long)Nn * 16, stream);  // mx + den (0u < f_ord(-inf): below-all sentinel)
    k_alpha<<<(EF + 255) / 256, 256, 0, stream>>>(
        ei, ea, ea_mean, deg, K8c, a_src, a_dst, exb, (unsigned*)mxf);
    k_expsum<<<(EF * 8 + 255) / 256, 256, 0, stream>>>(ei, exb, (const unsigned*)mxf, den);
    k_aggregate<<<(EF * 64 + 255) / 256, 256, 0, stream>>>(ei, xs, exb, den, gout);
    zero(bnsum, 512, stream);          // bnsum + bnss
    k_bn_stats<<<Nn / 128, 256, 0, stream>>>(gout, bnsum, bnss);
    k_bn_apply<<<(int)((NHF + 255) / 256), 256, 0, stream>>>(
        gout, bnsum, bnss, bn_g + i * Hh, bn_b + i * Hh, h, hbf, i > 0 ? 1 : 0);
  }

  // ---- MHA ----
  cvt(mha_in_w, wbf, (long)3 * Hh * Hh, stream);          // already [Nc,K]
  gemm(hbf, wbf, mha_in_b, qkv, nullptr, Nn, Hh, 3 * Hh, 0, stream);  // qkv [N,768]
  k_mha_attn<<<Gg * NHc, 128, 0, stream>>>(qkv, obuf, obf);
  cvt(mha_ow, wbf, (long)Hh * Hh, stream);                // already [Nc,K]
  gemm(obf, wbf, mha_ob, gout, nullptr, Nn, Hh, Hh, 0, stream);       // proj
  k_pool<<<Gg, 256, 0, stream>>>(gout, pooled, pbf);

  // ---- shared MLP + task heads ----
  cvtT(sh_w1, wbf, Hh, Hh, stream);
  gemm(pbf, wbf, sh_b1, s1, s1bf, Gg, Hh, Hh, 1, stream);
  cvtT(sh_w2, wbf, Hh, Hh / 2, stream);
  gemm(s1bf, wbf, sh_b2, s2, s2bf, Gg, Hh, Hh / 2, 1, stream);
  for (int t = 0; t < Tc; ++t) {
    cvtT(head_w1 + (size_t)t * (Hh / 2) * (Hh / 4), wbf, Hh / 2, Hh / 4, stream);
    gemm(s2bf, wbf, head_b1 + t * (Hh / 4), hh + (size_t)t * Gg * 64, nullptr,
         Gg, Hh / 2, Hh / 4, 1, stream);
  }
  k_head_final<<<(Gg * Tc + 255) / 256, 256, 0, stream>>>(hh, head_w2, head_b2, (float*)d_out);
}
